// DecoderRNN_56581899157814
// MI455X (gfx1250) — matrix operand order repaired and checked by reading, not compiled, not run on hardware
//
#include <hip/hip_runtime.h>
#include <hip/hip_bf16.h>

// ---------------------------------------------------------------------------
// Problem constants (match reference)
// ---------------------------------------------------------------------------
constexpr int H  = 1024;   // hidden
constexpr int V  = 512;    // vocab
constexpr int L  = 2;      // layers
constexpr int T  = 256;    // time steps
constexpr int B  = 256;    // batch
constexpr int FP = 2048;   // fingerprint dim

typedef __bf16 bf16_t;
typedef __attribute__((ext_vector_type(16))) __bf16 v16bf;
typedef __attribute__((ext_vector_type(8)))  __bf16 v8bf;
typedef __attribute__((ext_vector_type(8)))  float  v8f;

__device__ __forceinline__ bf16_t f2bf(float f) { return (bf16_t)f; }

// ---------------------------------------------------------------------------
// WMMA fragment loaders (bf16, 16x16x32), per CDNA5 ISA VGPR layouts.
// A (16x32, MxK): lane<16 -> M=lane, K chunks [kb..kb+7],[kb+16..kb+23], kb=k0
//                 lane>=16 -> M=lane-16, kb=k0+8.
// B (32x16, KxN): lane<16 -> N=lane, K chunks [kb..kb+7],[kb+16..kb+23], kb=k0;
//                 lane>=16 -> N=lane-16, kb=k0+8.  Column n of B == row n of W (row-major W).
// ---------------------------------------------------------------------------
__device__ __forceinline__ v16bf load_a_row(const bf16_t* __restrict__ rowp,
                                            int kb) {
  const bf16_t* p = rowp + kb;
  v8bf lo = *(const v8bf*)p;
  v8bf hi = *(const v8bf*)(p + 16);
  v16bf a;
#pragma unroll
  for (int i = 0; i < 8; ++i) { a[i] = lo[i]; a[i + 8] = hi[i]; }
  return a;
}

__device__ __forceinline__ v16bf load_b_frag(const bf16_t* __restrict__ wbase,
                                             int ld, int col0, int k0, int lane) {
  int r  = col0 + (lane & 15);                 // W row == output column
  int kb = k0 + ((lane >> 4) << 3);            // +8 for upper lane half
  const bf16_t* p = wbase + (size_t)r * ld + kb;
  v8bf lo = *(const v8bf*)p;
  v8bf hi = *(const v8bf*)(p + 16);
  v16bf b;
#pragma unroll
  for (int i = 0; i < 8; ++i) { b[i] = lo[i]; b[i + 8] = hi[i]; }
  return b;
}

__device__ __forceinline__ v8f wmma_bf16(v16bf a, v16bf b, v8f c) {
  return __builtin_amdgcn_wmma_f32_16x16x32_bf16(false, a, false, b,
                                                 (short)0, c, false, false);
}

__device__ __forceinline__ float sigmoidf_(float x) {
  return 1.0f / (1.0f + __expf(-x));
}

// ---------------------------------------------------------------------------
// fp32 -> bf16 conversion (optionally with relu, for the embedding table)
// ---------------------------------------------------------------------------
__global__ void cvt_bf16_kernel(const float* __restrict__ src,
                                bf16_t* __restrict__ dst, int n) {
  int i = blockIdx.x * blockDim.x + threadIdx.x;
  if (i < n) dst[i] = f2bf(src[i]);
}

__global__ void cvt_relu_bf16_kernel(const float* __restrict__ src,
                                     bf16_t* __restrict__ dst, int n) {
  int i = blockIdx.x * blockDim.x + threadIdx.x;
  if (i < n) { float v = src[i]; dst[i] = f2bf(v > 0.0f ? v : 0.0f); }
}

// ---------------------------------------------------------------------------
// h0 = relu(fingerprints @ Wc^T + bc), broadcast to both layers.
// M=B=256, N=H=1024, K=FP=2048.  One wave per 16x16 tile (one-time cost).
// ---------------------------------------------------------------------------
__global__ __launch_bounds__(32) void connection_kernel(
    const bf16_t* __restrict__ fp_bf,   // [B,FP]
    const bf16_t* __restrict__ Wc_bf,   // [H,FP]
    const float*  __restrict__ bc,      // [H]
    float*  __restrict__ h_f32,         // [L,B,H] f32 buffer 0
    bf16_t* __restrict__ h0_bf,         // [B,H] layer-0 bf16 init
    bf16_t* __restrict__ h1_bf)         // [B,H] layer-1 bf16 init
{
  const int lane = threadIdx.x;
  const int tm = blockIdx.x >> 6;       // 16 M tiles
  const int tn = blockIdx.x & 63;       // 64 N tiles
  const int m0 = tm * 16, n0 = tn * 16;

  const bf16_t* arow = fp_bf + (size_t)(m0 + (lane & 15)) * FP;
  const int kba = ((lane >> 4) << 3);

  v8f acc = {};
  for (int k0 = 0; k0 < FP; k0 += 32) {
    v16bf a = load_a_row(arow, k0 + kba);
    v16bf b = load_b_frag(Wc_bf, FP, n0, k0, lane);
    acc = wmma_bf16(a, b, acc);
  }
  const int n = n0 + (lane & 15);
  const float bias = bc[n];
  const int mbase = m0 + ((lane >> 4) << 3);
#pragma unroll
  for (int r = 0; r < 8; ++r) {
    int m = mbase + r;
    float v = acc[r] + bias;
    v = v > 0.0f ? v : 0.0f;
    size_t idx = (size_t)m * H + n;
    h_f32[idx]                 = v;   // layer 0
    h_f32[(size_t)B * H + idx] = v;   // layer 1
    h0_bf[idx] = f2bf(v);
    h1_bf[idx] = f2bf(v);
  }
}

// ---------------------------------------------------------------------------
// Fused GRU layer step.  One wave owns TWO 16x16 h-tiles (same N, M and M+16)
// so every weight B-fragment is reused twice; A reused across the 3 gates.
// GATHER=true: layer-0 input rows come straight from relu(emb)[token_t]
// (embedding lookup fused into the A-fragment gather; no x staging kernel).
// ---------------------------------------------------------------------------
template <bool GATHER>
__global__ __launch_bounds__(32) void gru_layer_kernel(
    const bf16_t* __restrict__ xsrc,    // GATHER ? relu(emb) [V,H] : input [B,H]
    const int*    __restrict__ target,  // [B,T] (used when GATHER)
    int t,
    const bf16_t* __restrict__ h_bf,    // [B,H] old h (bf16)
    const float*  __restrict__ h_f32,   // [B,H] old h (f32)
    const bf16_t* __restrict__ Wih,     // [3H,H]
    const bf16_t* __restrict__ Whh,     // [3H,H]
    const float*  __restrict__ bih,     // [3H]
    const float*  __restrict__ bhh,     // [3H]
    float*  __restrict__ hn_f32,        // [B,H] new h (f32)
    bf16_t* __restrict__ hn_bf)         // [B,H] new h (bf16)
{
  const int lane = threadIdx.x;
  const int mg = blockIdx.x >> 6;       // 8 M groups (2 tiles each)
  const int tn = blockIdx.x & 63;       // 64 N tiles (H/16)
  const int n0 = tn * 16;
  const int m0 = mg * 32;

  const int rlo  = lane & 15;
  const int row0 = m0 + rlo;
  const int row1 = m0 + 16 + rlo;
  const bf16_t *xrow0, *xrow1;
  if (GATHER) {
    int tok0 = (t == 0) ? 0 : target[row0 * T + t - 1];
    int tok1 = (t == 0) ? 0 : target[row1 * T + t - 1];
    xrow0 = xsrc + (size_t)tok0 * H;
    xrow1 = xsrc + (size_t)tok1 * H;
  } else {
    xrow0 = xsrc + (size_t)row0 * H;
    xrow1 = xsrc + (size_t)row1 * H;
  }
  const bf16_t* hrow0 = h_bf + (size_t)row0 * H;
  const bf16_t* hrow1 = h_bf + (size_t)row1 * H;
  const int kba = ((lane >> 4) << 3);

  v8f acc[12] = {};   // [mtile 0: ir,iz,in,hr,hz,hn | mtile 1: same]
  for (int k0 = 0; k0 < H; k0 += 32) {
    v16bf ax0 = load_a_row(xrow0, k0 + kba);
    v16bf ax1 = load_a_row(xrow1, k0 + kba);
    v16bf ah0 = load_a_row(hrow0, k0 + kba);
    v16bf ah1 = load_a_row(hrow1, k0 + kba);
#pragma unroll
    for (int g = 0; g < 3; ++g) {
      v16bf bi = load_b_frag(Wih, H, g * H + n0, k0, lane);
      acc[g]     = wmma_bf16(ax0, bi, acc[g]);
      acc[6 + g] = wmma_bf16(ax1, bi, acc[6 + g]);
      v16bf bh = load_b_frag(Whh, H, g * H + n0, k0, lane);
      acc[3 + g] = wmma_bf16(ah0, bh, acc[3 + g]);
      acc[9 + g] = wmma_bf16(ah1, bh, acc[9 + g]);
    }
  }

  const int n = n0 + (lane & 15);
  const float bir = bih[0 * H + n], biz = bih[1 * H + n], bin_ = bih[2 * H + n];
  const float bhr = bhh[0 * H + n], bhz = bhh[1 * H + n], bhn_ = bhh[2 * H + n];
#pragma unroll
  for (int mt = 0; mt < 2; ++mt) {
    const int mbase = m0 + mt * 16 + ((lane >> 4) << 3);
    const int a0 = mt * 6;
#pragma unroll
    for (int r = 0; r < 8; ++r) {
      int m = mbase + r;
      float ir = acc[a0 + 0][r] + bir, iz = acc[a0 + 1][r] + biz,
            in_ = acc[a0 + 2][r] + bin_;
      float hr = acc[a0 + 3][r] + bhr, hz = acc[a0 + 4][r] + bhz,
            hn = acc[a0 + 5][r] + bhn_;
      float rg = sigmoidf_(ir + hr);
      float zg = sigmoidf_(iz + hz);
      float ng = tanhf(in_ + rg * hn);
      float hold = h_f32[(size_t)m * H + n];
      float hnew = (1.0f - zg) * ng + zg * hold;
      hn_f32[(size_t)m * H + n] = hnew;
      hn_bf [(size_t)m * H + n] = f2bf(hnew);
    }
  }
}

// ---------------------------------------------------------------------------
// Fused logits GEMM + log_softmax.  One workgroup (16 waves, 512 thr) owns 16
// rows: each wave computes two 16x16 WMMA tiles (V = 32 N-tiles), stages the
// [16,512] logits block in LDS, then each wave log-softmaxes one row with
// __shfl_xor reductions and writes log-probs to d_out.
//   t_fixed >= 0 : rows are batch rows b at step t_fixed (fallback path)
//   t_fixed <  0 : rows index h1 history [T,B,H]; g = t*B + b (batched path)
// ---------------------------------------------------------------------------
__global__ __launch_bounds__(512) void logits_lsm_kernel(
    const bf16_t* __restrict__ h1,      // [rows, H]
    const bf16_t* __restrict__ Wo_bf,   // [V,H]
    const float*  __restrict__ bo,      // [V]
    float* __restrict__ out,            // [B,T,V]
    int t_fixed)
{
  __shared__ float lds[16][V];          // 32 KB
  const int lane = threadIdx.x & 31;
  const int wave = threadIdx.x >> 5;    // 0..15
  const int g0 = blockIdx.x * 16;

  const bf16_t* arow = h1 + (size_t)(g0 + (lane & 15)) * H;
  const int kba = ((lane >> 4) << 3);
  const int n0 = wave * 32;             // this wave's two N-tiles

  v8f acc0 = {}, acc1 = {};
  for (int k0 = 0; k0 < H; k0 += 32) {
    v16bf a  = load_a_row(arow, k0 + kba);
    v16bf b0 = load_b_frag(Wo_bf, H, n0,      k0, lane);
    v16bf b1 = load_b_frag(Wo_bf, H, n0 + 16, k0, lane);
    acc0 = wmma_bf16(a, b0, acc0);
    acc1 = wmma_bf16(a, b1, acc1);
  }
  const int nl = lane & 15;
  const float bias0 = bo[n0 + nl], bias1 = bo[n0 + 16 + nl];
  const int mb = ((lane >> 4) << 3);
#pragma unroll
  for (int r = 0; r < 8; ++r) {
    lds[mb + r][n0 + nl]      = acc0[r] + bias0;
    lds[mb + r][n0 + 16 + nl] = acc1[r] + bias1;
  }
  __syncthreads();

  // log_softmax: wave w owns row w
  float* row = lds[wave];
  float mx = -3.402823e38f;
#pragma unroll
  for (int i = lane; i < V; i += 32) mx = fmaxf(mx, row[i]);
#pragma unroll
  for (int off = 16; off; off >>= 1) mx = fmaxf(mx, __shfl_xor(mx, off, 32));
  float s = 0.0f;
#pragma unroll
  for (int i = lane; i < V; i += 32) s += __expf(row[i] - mx);
#pragma unroll
  for (int off = 16; off; off >>= 1) s += __shfl_xor(s, off, 32);
  float lse = mx + __logf(s);

  const int g = g0 + wave;
  int b, tt;
  if (t_fixed >= 0) { b = g; tt = t_fixed; }
  else              { tt = g >> 8; b = g & (B - 1); }   // g = t*B + b, B=256
  float* orow = out + ((size_t)b * T + tt) * V;
#pragma unroll
  for (int i = lane; i < V; i += 32) orow[i] = row[i] - lse;
}

// ---------------------------------------------------------------------------
// copy final hidden state into d_out tail
// ---------------------------------------------------------------------------
__global__ void copy_f32_kernel(const float* __restrict__ src,
                                float* __restrict__ dst, int n) {
  int i = blockIdx.x * blockDim.x + threadIdx.x;
  if (i < n) dst[i] = src[i];
}

// ---------------------------------------------------------------------------
// Host-side orchestration
// ---------------------------------------------------------------------------
extern "C" void kernel_launch(void* const* d_in, const int* in_sizes, int n_in,
                              void* d_out, int out_size, void* d_ws, size_t ws_size,
                              hipStream_t stream) {
  (void)in_sizes; (void)n_in; (void)out_size;

  const float* fingerprints = (const float*)d_in[1];   // [B,FP]
  const int*   target       = (const int*)  d_in[2];   // [B,T]
  const float* emb          = (const float*)d_in[3];   // [V,H]
  const float* Wc           = (const float*)d_in[4];   // [H,FP]
  const float* bc           = (const float*)d_in[5];   // [H]
  const float* Wih          = (const float*)d_in[6];   // [L,3H,H]
  const float* Whh          = (const float*)d_in[7];   // [L,3H,H]
  const float* bih          = (const float*)d_in[8];   // [L,3H]
  const float* bhh          = (const float*)d_in[9];   // [L,3H]
  const float* Wo           = (const float*)d_in[10];  // [V,H]
  const float* bo           = (const float*)d_in[11];  // [V]

  float* out     = (float*)d_out;                      // [B,T,V]
  float* h_final = out + (size_t)B * T * V;            // [L,B,H]

  // ---- workspace carve-up -------------------------------------------------
  const size_t nWc  = (size_t)H * FP;
  const size_t nWih = (size_t)L * 3 * H * H;
  const size_t nWo  = (size_t)V * H;
  const size_t nFp  = (size_t)B * FP;
  const size_t nEmb = (size_t)V * H;
  const size_t BH   = (size_t)B * H;
  const size_t nHf  = (size_t)L * B * H;               // per f32 buffer

  bf16_t* p = (bf16_t*)d_ws;
  bf16_t* Wc_bf   = p; p += nWc;
  bf16_t* Wih_bf  = p; p += nWih;
  bf16_t* Whh_bf  = p; p += nWih;
  bf16_t* Wo_bf   = p; p += nWo;
  bf16_t* fp_bf   = p; p += nFp;
  bf16_t* embr_bf = p; p += nEmb;       // relu(emb) in bf16
  bf16_t* h0bf_a  = p; p += BH;         // layer-0 bf16 double buffer
  bf16_t* h0bf_b  = p; p += BH;
  bf16_t* h1bf_i  = p; p += BH;         // layer-1 bf16 init
  bf16_t* h1sl_a  = p; p += BH;         // layer-1 bf16 slots (fallback)
  bf16_t* h1sl_b  = p; p += BH;
  float*  hf0     = (float*)p;          // f32 h double buffers [L,B,H]
  float*  hf1     = hf0 + nHf;
  bf16_t* h1hist  = (bf16_t*)(hf1 + nHf);  // [T,B,H] bf16 history (if it fits)

  const size_t base_bytes = (size_t)((char*)h1hist - (char*)d_ws);
  const size_t hist_bytes = (size_t)T * B * H * sizeof(bf16_t);
  const bool   batched    = (base_bytes + hist_bytes) <= ws_size;

  const int CT = 256;
  auto blocks = [](size_t n, int t) { return (int)((n + t - 1) / t); };

  // ---- one-time bf16 conversions -----------------------------------------
  cvt_bf16_kernel<<<blocks(nWc,  CT), CT, 0, stream>>>(Wc,  Wc_bf,  (int)nWc);
  cvt_bf16_kernel<<<blocks(nWih, CT), CT, 0, stream>>>(Wih, Wih_bf, (int)nWih);
  cvt_bf16_kernel<<<blocks(nWih, CT), CT, 0, stream>>>(Whh, Whh_bf, (int)nWih);
  cvt_bf16_kernel<<<blocks(nWo,  CT), CT, 0, stream>>>(Wo,  Wo_bf,  (int)nWo);
  cvt_bf16_kernel<<<blocks(nFp,  CT), CT, 0, stream>>>(fingerprints, fp_bf, (int)nFp);
  cvt_relu_bf16_kernel<<<blocks(nEmb, CT), CT, 0, stream>>>(emb, embr_bf, (int)nEmb);

  // ---- h0 = relu(connection(fingerprints)) --------------------------------
  connection_kernel<<<16 * 64, 32, 0, stream>>>(fp_bf, Wc_bf, bc,
                                                hf0, h0bf_a, h1bf_i);

  const size_t W3H = (size_t)3 * H * H;

  // ---- sequential decode ---------------------------------------------------
  for (int t = 0; t < T; ++t) {
    float*  hf_cur  = (t & 1) ? hf1 : hf0;
    float*  hf_nxt  = (t & 1) ? hf0 : hf1;
    bf16_t* h0_cur  = (t & 1) ? h0bf_b : h0bf_a;
    bf16_t* h0_nxt  = (t & 1) ? h0bf_a : h0bf_b;

    bf16_t* h1_prev, *h1_next;
    if (batched) {
      h1_prev = (t == 0) ? h1bf_i : (h1hist + (size_t)(t - 1) * BH);
      h1_next = h1hist + (size_t)t * BH;
    } else {
      h1_prev = (t == 0) ? h1bf_i : ((t & 1) ? h1sl_a : h1sl_b);
      h1_next = (t & 1) ? h1sl_b : h1sl_a;
    }

    // layer 0: input rows gathered from relu(emb)[token_t] (fused embedding)
    gru_layer_kernel<true><<<8 * 64, 32, 0, stream>>>(
        embr_bf, target, t, h0_cur, hf_cur,
        Wih_bf, Whh_bf, bih, bhh,
        hf_nxt, h0_nxt);

    // layer 1: input = new h of layer 0
    gru_layer_kernel<false><<<8 * 64, 32, 0, stream>>>(
        h0_nxt, target, t, h1_prev, hf_cur + BH,
        Wih_bf + W3H, Whh_bf + W3H, bih + 3 * H, bhh + 3 * H,
        hf_nxt + BH, h1_next);

    if (!batched) {
      logits_lsm_kernel<<<B / 16, 512, 0, stream>>>(h1_next, Wo_bf, bo, out, t);
    }
  }

  if (batched) {
    // one big parallel pass: [T*B, V] = h1_hist @ Wo^T + bo, then log_softmax
    logits_lsm_kernel<<<(T * B) / 16, 512, 0, stream>>>(h1hist, Wo_bf, bo, out, -1);
  }

  // T even -> final f32 h resides in buffer 0
  copy_f32_kernel<<<blocks(nHf, CT), CT, 0, stream>>>(hf0, h_final, (int)nHf);
}
